// TnlAttention_34522947125495
// MI455X (gfx1250) — matrix operand order repaired and checked by reading, not compiled, not run on hardware
//
#include <hip/hip_runtime.h>

typedef __bf16 bf16_t;
typedef __attribute__((ext_vector_type(16))) __bf16 v16bf;
typedef __attribute__((ext_vector_type(8)))  float  v8f;
typedef __attribute__((ext_vector_type(4)))  unsigned int v4u;
typedef __attribute__((ext_vector_type(8)))  int v8i;
typedef __attribute__((ext_vector_type(4)))  int v4i;

// ---------------------------------------------------------------------------
// CDNA5 WMMA fragment helpers (wave32). LDS tiles are stored:
//   A: [M][K] row-major (K contiguous), B: [N][K] (i.e. B transposed).
// A 16x32 bf16 layout: lane<16: M=lane, K in {0..7,16..23}; lane>=16: M=lane-16,
// K in {8..15,24..31}. B 32x16: lane<16: N=lane, K in {0..7,16..23}; lane>=16: K in {8..15,24..31}.
// ---------------------------------------------------------------------------
static __device__ __forceinline__ v16bf ld_frag_a(const bf16_t* tile, int stride, int lane) {
    const int m = lane & 15, hh = lane >> 4;
    const bf16_t* p = tile + m * stride + hh * 8;
    union { v16bf f; uint2 u[4]; } r;
    r.u[0] = *(const uint2*)(p);
    r.u[1] = *(const uint2*)(p + 4);
    r.u[2] = *(const uint2*)(p + 16);
    r.u[3] = *(const uint2*)(p + 20);
    return r.f;
}

static __device__ __forceinline__ v16bf ld_frag_b(const bf16_t* tile, int stride, int lane) {
    const int n = lane & 15, hh = lane >> 4;
    const bf16_t* p = tile + n * stride + hh * 8;
    union { v16bf f; uint2 u[4]; } r;
    r.u[0] = *(const uint2*)(p);
    r.u[1] = *(const uint2*)(p + 4);
    r.u[2] = *(const uint2*)(p + 16);
    r.u[3] = *(const uint2*)(p + 20);
    return r.f;
}

static __device__ __forceinline__ v8f wmma_bf16(v16bf a, v16bf b, v8f c) {
    return __builtin_amdgcn_wmma_f32_16x16x32_bf16(false, a, false, b, (short)0, c, false, false);
}

static __device__ __forceinline__ v16bf frag_scale(v16bf f, float sc) {
    v16bf r;
#pragma unroll
    for (int j = 0; j < 16; ++j) r[j] = (bf16_t)(sc * (float)f[j]);
    return r;
}

// LDS byte offset: flat LDS addresses keep the offset in addr[31:0] (ISA 10.2)
static __device__ __forceinline__ unsigned lds_off(const void* p) {
    return (unsigned)(unsigned long long)p;
}

// Async global->LDS 16B copy per lane (ASYNCcnt tracked)
static __device__ __forceinline__ void async_copy_b128(unsigned lds, const void* gp) {
    asm volatile("global_load_async_to_lds_b128 %0, %1, off"
                 :: "v"(lds), "v"(gp) : "memory");
}
static __device__ __forceinline__ void wait_asynccnt0() {
    asm volatile("s_wait_asynccnt 0x0" ::: "memory");
}

// ---------------------------------------------------------------------------
// TDM: load a 128x64 bf16 tile (row stride 1024 elems) into LDS with 16-elem
// padding after each 64-elem row => LDS row pitch 80 elements (160 B).
// D# group layout per CDNA5 ISA.
// ---------------------------------------------------------------------------
static __device__ __forceinline__ void tdm_load_tile_128x64(const bf16_t* g, unsigned lds) {
    unsigned long long ga = (unsigned long long)g;
    v4u g0;
    g0[0] = 1u;                                            // count=1, user desc
    g0[1] = lds;                                           // lds_addr (bytes)
    g0[2] = (unsigned)ga;                                  // global_addr[31:0]
    g0[3] = (unsigned)((ga >> 32) & 0x01FFFFFFu) | 0x80000000u; // addr[56:32] | type=2
    v8i g1;
    g1[0] = (1 << 16)            // data_size = 2 bytes
          | (1 << 20)            // pad_enable
          | (4 << 22)            // pad_interval: 32 DWORDs (one 64-elem row)
          | (7 << 25);           // pad_amount: 8 DWORDs (16 elems)
    g1[1] = (64u & 0xFFFFu) << 16;         // tensor_dim0[15:0] @ bits 63:48
    g1[2] = (int)(128u << 16);             // tensor_dim0[31:16]=0 | tensor_dim1[15:0]
    g1[3] = (int)(64u << 16);              // tensor_dim1[31:16]=0 | tile_dim0=64
    g1[4] = 128;                           // tile_dim1=128 | tile_dim2=0
    g1[5] = 1024;                          // tensor_dim0_stride[31:0] = E
    g1[6] = 0;                             // stride0[47:32]=0 | stride1 lo=0
    g1[7] = 0;
    v4i g2 = {0, 0, 0, 0};
    v4i g3 = {0, 0, 0, 0};
    asm volatile("tensor_load_to_lds %0, %1, %2, %3"
                 :: "s"(g0), "s"(g1), "s"(g2), "s"(g3) : "memory");
}

// ---------------------------------------------------------------------------
// fp32 -> bf16 conversion
// ---------------------------------------------------------------------------
__global__ void tnl_cvt_bf16(const float* __restrict__ s, bf16_t* __restrict__ d, int n) {
    int i = blockIdx.x * 256 + threadIdx.x;
    if (i < n) d[i] = (bf16_t)s[i];
}

// ---------------------------------------------------------------------------
// Tiled bf16 GEMM: C[M,N] = act(A[M,K] @ B[K,N]).  BM=128, BK=32.
// 256 threads = 8 waves; wave w computes m-tile row w (16 rows x BN cols).
// A tile staged with async global->LDS b128 copies. ACT: 0 none,1 silu,2 sigm.
// ---------------------------------------------------------------------------
template <int BN, int ACT, typename OutT>
__global__ __launch_bounds__(256) void tnl_gemm_bf16(const bf16_t* __restrict__ A,
                                                     const bf16_t* __restrict__ Bm,
                                                     OutT* __restrict__ Cout,
                                                     int M, int N, int K) {
    constexpr int BM = 128, BK = 32, LDA = BK + 8, NT = BN / 16;   // 80B rows (16B aligned)
    __shared__ bf16_t As[BM * LDA];
    __shared__ bf16_t Bs[BN * LDA];
    const int tid = threadIdx.x, lane = tid & 31, w = tid >> 5;
    const int m0 = blockIdx.y * BM, n0 = blockIdx.x * BN;

    v8f acc[NT];
#pragma unroll
    for (int nt = 0; nt < NT; ++nt)
#pragma unroll
        for (int r = 0; r < 8; ++r) acc[nt][r] = 0.f;

    for (int k0 = 0; k0 < K; k0 += BK) {
        // A tile: [BM][BK] row-major via async 16B copies (512 chunks / 256 thr)
        for (int idx = tid; idx < BM * (BK / 8); idx += 256) {
            int r = idx >> 2;
            int c8 = (idx & 3) * 8;
            async_copy_b128(lds_off(&As[r * LDA + c8]),
                            A + (size_t)(m0 + r) * K + k0 + c8);
        }
        // B tile: transpose global [K][N] -> LDS [N][K]
        for (int idx = tid; idx < BK * (BN / 4); idx += 256) {
            int kk = idx / (BN / 4);
            int n4 = (idx % (BN / 4)) * 4;
            union { uint2 u; bf16_t e[4]; } d;
            d.u = *(const uint2*)(Bm + (size_t)(k0 + kk) * N + n0 + n4);
            Bs[(n4 + 0) * LDA + kk] = d.e[0];
            Bs[(n4 + 1) * LDA + kk] = d.e[1];
            Bs[(n4 + 2) * LDA + kk] = d.e[2];
            Bs[(n4 + 3) * LDA + kk] = d.e[3];
        }
        wait_asynccnt0();
        __syncthreads();
        v16bf a = ld_frag_a(&As[(w * 16) * LDA], LDA, lane);
#pragma unroll
        for (int nt = 0; nt < NT; ++nt) {
            v16bf b = ld_frag_b(&Bs[(nt * 16) * LDA], LDA, lane);
            acc[nt] = wmma_bf16(a, b, acc[nt]);
        }
        __syncthreads();
    }

#pragma unroll
    for (int nt = 0; nt < NT; ++nt) {
#pragma unroll
        for (int r = 0; r < 8; ++r) {
            int m = m0 + w * 16 + r + ((lane >> 4) << 3);
            int n = n0 + nt * 16 + (lane & 15);
            float v = acc[nt][r];
            if constexpr (ACT == 1) v = v / (1.f + __expf(-v));          // silu
            else if constexpr (ACT == 2) v = 1.f / (1.f + __expf(-v));   // sigmoid
            if constexpr (__is_same(OutT, float)) Cout[(size_t)m * N + n] = v;
            else                                  Cout[(size_t)m * N + n] = (bf16_t)v;
        }
    }
}

// ---------------------------------------------------------------------------
// Lightning attention: one workgroup per (h, b); sequential over 64 chunks.
// C=128, D=64, E=1024, N=8192.  Q/K chunks staged via TDM tensor_load_to_lds.
// ---------------------------------------------------------------------------
__global__ __launch_bounds__(256) void tnl_attn(const bf16_t* __restrict__ qb,
                                                const bf16_t* __restrict__ kb,
                                                const bf16_t* __restrict__ vb,
                                                bf16_t* __restrict__ ob) {
    constexpr int C = 128, D = 64, E = 1024, N = 8192;
    constexpr int LQ = D + 16;  // 80 elems = 160B rows (matches TDM padding)
    constexpr int LC = C + 8;   // Kt/Vt/Ss row stride

    __shared__ bf16_t Qs[C * LQ];
    __shared__ bf16_t Ks[C * LQ];
    __shared__ bf16_t Kt[D * LC];    // K^T scaled by lam_k  (A for KV update)
    __shared__ bf16_t Vt[D * LC];    // V^T                  (B for S@V and KV update)
    __shared__ bf16_t Ss[C * LC];    // masked scores        (A for S@V)
    __shared__ bf16_t KVt[D * LQ];   // KV^T bf16            (B for Q@KV)
    __shared__ float  KVf[D * D];    // KV state f32

    const int h = blockIdx.x, b = blockIdx.y;
    const int tid = threadIdx.x, lane = tid & 31, w = tid >> 5;

    const float s = exp2f(-0.5f * (float)(h + 1)) * (1.0f + 1e-5f);
    const float lam_c = __expf(-s * (float)C);

    for (int idx = tid; idx < D * D; idx += 256) KVf[idx] = 0.f;
    for (int idx = tid; idx < D * LQ; idx += 256) KVt[idx] = (bf16_t)0.f;
    __syncthreads();

    for (int c = 0; c < N / C; ++c) {
        const size_t base = (size_t)b * N + (size_t)c * C;

        // ---- stage chunk: Q,K via TDM (wave 0 issues, TENSORcnt tracked) ----
        if (w == 0) {
            tdm_load_tile_128x64(qb + base * E + h * D, lds_off(&Qs[0]));
            tdm_load_tile_128x64(kb + base * E + h * D, lds_off(&Ks[0]));
            __builtin_amdgcn_s_wait_tensorcnt(0);
        }
        for (int idx = tid; idx < C * (D / 4); idx += 256) {       // V^T
            int j = idx >> 4, d4 = (idx & 15) * 4;
            union { uint2 u; bf16_t e[4]; } d;
            d.u = *(const uint2*)(vb + (base + j) * E + h * D + d4);
            Vt[(d4 + 0) * LC + j] = d.e[0];
            Vt[(d4 + 1) * LC + j] = d.e[1];
            Vt[(d4 + 2) * LC + j] = d.e[2];
            Vt[(d4 + 3) * LC + j] = d.e[3];
        }
        for (int idx = tid; idx < C * (D / 4); idx += 256) {       // K^T * lam_k
            int j = idx >> 4, d4 = (idx & 15) * 4;
            float lamk = __expf(-s * (float)(C - 1 - j));
            union { uint2 u; bf16_t e[4]; } d;
            d.u = *(const uint2*)(kb + (base + j) * E + h * D + d4);
            Kt[(d4 + 0) * LC + j] = (bf16_t)((float)d.e[0] * lamk);
            Kt[(d4 + 1) * LC + j] = (bf16_t)((float)d.e[1] * lamk);
            Kt[(d4 + 2) * LC + j] = (bf16_t)((float)d.e[2] * lamk);
            Kt[(d4 + 3) * LC + j] = (bf16_t)((float)d.e[3] * lamk);
        }
        __syncthreads();

        // ---- phase 1: masked scores S = mask .* (Q @ K^T), wave w -> m-tile w ----
        {
            v16bf a0 = ld_frag_a(&Qs[(w * 16) * LQ], LQ, lane);
            v16bf a1 = ld_frag_a(&Qs[(w * 16) * LQ + 32], LQ, lane);
#pragma unroll
            for (int nt = 0; nt < 8; ++nt) {
                v16bf b0 = ld_frag_b(&Ks[(nt * 16) * LQ], LQ, lane);
                v16bf b1 = ld_frag_b(&Ks[(nt * 16) * LQ + 32], LQ, lane);
                v8f acc;
#pragma unroll
                for (int r = 0; r < 8; ++r) acc[r] = 0.f;
                acc = wmma_bf16(a0, b0, acc);
                acc = wmma_bf16(a1, b1, acc);
#pragma unroll
                for (int r = 0; r < 8; ++r) {
                    int i = w * 16 + r + ((lane >> 4) << 3);
                    int j = nt * 16 + (lane & 15);
                    float mv = (i >= j) ? __expf(-s * (float)(i - j)) : 0.f;
                    Ss[i * LC + j] = (bf16_t)(acc[r] * mv);
                }
            }
        }
        __syncthreads();

        // ---- phase 2: O = S @ V + lam_q .* (Q @ KV) ----
        {
            v16bf aS[4];
#pragma unroll
            for (int ks = 0; ks < 4; ++ks) aS[ks] = ld_frag_a(&Ss[(w * 16) * LC + ks * 32], LC, lane);
            const int i_lane = w * 16 + (lane & 15);
            const float lamq = __expf(-s * (float)(i_lane + 1));
            v16bf aQ[2];
#pragma unroll
            for (int ks = 0; ks < 2; ++ks)
                aQ[ks] = frag_scale(ld_frag_a(&Qs[(w * 16) * LQ + ks * 32], LQ, lane), lamq);
#pragma unroll
            for (int nt = 0; nt < 4; ++nt) {
                v8f acc;
#pragma unroll
                for (int r = 0; r < 8; ++r) acc[r] = 0.f;
#pragma unroll
                for (int ks = 0; ks < 4; ++ks)
                    acc = wmma_bf16(aS[ks], ld_frag_b(&Vt[(nt * 16) * LC + ks * 32], LC, lane), acc);
#pragma unroll
                for (int ks = 0; ks < 2; ++ks)
                    acc = wmma_bf16(aQ[ks], ld_frag_b(&KVt[(nt * 16) * LQ + ks * 32], LQ, lane), acc);
#pragma unroll
                for (int r = 0; r < 8; ++r) {
                    int i = w * 16 + r + ((lane >> 4) << 3);
                    int d = nt * 16 + (lane & 15);
                    ob[(base + i) * E + h * D + d] = (bf16_t)acc[r];
                }
            }
        }
        __syncthreads();

        // ---- phase 3: KV = lam_c*KV + (lam_k.*K)^T @ V ;  16 tiles over 8 waves ----
#pragma unroll
        for (int t2 = 0; t2 < 2; ++t2) {
            int t = w * 2 + t2;
            int mt = t >> 2, nt = t & 3;
            v8f acc;
#pragma unroll
            for (int r = 0; r < 8; ++r) acc[r] = 0.f;
#pragma unroll
            for (int ks = 0; ks < 4; ++ks) {
                v16bf a = ld_frag_a(&Kt[(mt * 16) * LC + ks * 32], LC, lane);
                v16bf bB = ld_frag_b(&Vt[(nt * 16) * LC + ks * 32], LC, lane);
                acc = wmma_bf16(a, bB, acc);
            }
#pragma unroll
            for (int r = 0; r < 8; ++r) {
                int d = mt * 16 + r + ((lane >> 4) << 3);
                int e = nt * 16 + (lane & 15);
                KVf[d * D + e] = lam_c * KVf[d * D + e] + acc[r];
            }
        }
        __syncthreads();
        for (int idx = tid; idx < D * D; idx += 256) {
            int d = idx >> 6, e = idx & 63;
            KVt[e * LQ + d] = (bf16_t)KVf[d * D + e];
        }
        __syncthreads();
    }
}

// ---------------------------------------------------------------------------
// LayerNorm(o)*gamma+beta, times gate (already sigmoid'ed) -> bf16 h
// ---------------------------------------------------------------------------
__global__ __launch_bounds__(256) void tnl_ln_gate(const bf16_t* __restrict__ ob,
                                                   const float* __restrict__ gamma,
                                                   const float* __restrict__ beta,
                                                   const bf16_t* __restrict__ gate,
                                                   bf16_t* __restrict__ hb) {
    const int row = blockIdx.x, tid = threadIdx.x;
    const size_t roff = (size_t)row * 1024;
    float vals[4], sum = 0.f, sq = 0.f;
#pragma unroll
    for (int j = 0; j < 4; ++j) {
        float v = (float)ob[roff + tid + j * 256];
        vals[j] = v; sum += v; sq += v * v;
    }
#pragma unroll
    for (int off = 16; off > 0; off >>= 1) {
        sum += __shfl_xor(sum, off, 32);
        sq  += __shfl_xor(sq,  off, 32);
    }
    __shared__ float s1[8], s2[8];
    if ((tid & 31) == 0) { s1[tid >> 5] = sum; s2[tid >> 5] = sq; }
    __syncthreads();
    sum = 0.f; sq = 0.f;
#pragma unroll
    for (int k = 0; k < 8; ++k) { sum += s1[k]; sq += s2[k]; }
    const float mu = sum * (1.f / 1024.f);
    const float var = sq * (1.f / 1024.f) - mu * mu;
    const float rstd = rsqrtf(var + 1e-5f);
#pragma unroll
    for (int j = 0; j < 4; ++j) {
        int e = tid + j * 256;
        float y = (vals[j] - mu) * rstd * gamma[e] + beta[e];
        y *= (float)gate[roff + e];
        hb[roff + e] = (bf16_t)y;
    }
}

// ---------------------------------------------------------------------------
// Launch: x,Wq,Wk,Wv,Wo,gamma,beta,Wg1,Wg2  ->  out fp32 [2,8192,1024]
// ---------------------------------------------------------------------------
extern "C" void kernel_launch(void* const* d_in, const int* in_sizes, int n_in,
                              void* d_out, int out_size, void* d_ws, size_t ws_size,
                              hipStream_t stream) {
    (void)in_sizes; (void)n_in; (void)out_size; (void)ws_size;
    const float* x     = (const float*)d_in[0];
    const float* Wq    = (const float*)d_in[1];
    const float* Wk    = (const float*)d_in[2];
    const float* Wv    = (const float*)d_in[3];
    const float* Wo    = (const float*)d_in[4];
    const float* gamma = (const float*)d_in[5];
    const float* beta  = (const float*)d_in[6];
    const float* Wg1   = (const float*)d_in[7];
    const float* Wg2   = (const float*)d_in[8];
    float* out = (float*)d_out;

    const int M = 2 * 8192;          // 16384 rows
    const int E = 1024, D = 64;
    const size_t MNE = (size_t)M * E;    // 16,777,216
    const size_t EE  = (size_t)E * E;    // 1,048,576

    char* ws = (char*)d_ws;
    size_t off = 0;
    auto take = [&](size_t bytes) { char* p = ws + off; off = (off + bytes + 255) & ~(size_t)255; return p; };
    bf16_t* xb   = (bf16_t*)take(MNE * 2);          // also reused as hb
    bf16_t* qb   = (bf16_t*)take(MNE * 2);          // also reused as gate
    bf16_t* kb   = (bf16_t*)take(MNE * 2);
    bf16_t* vb   = (bf16_t*)take(MNE * 2);
    bf16_t* obuf = (bf16_t*)take(MNE * 2);
    bf16_t* wqb  = (bf16_t*)take(EE * 2);
    bf16_t* wkb  = (bf16_t*)take(EE * 2);
    bf16_t* wvb  = (bf16_t*)take(EE * 2);
    bf16_t* wob  = (bf16_t*)take(EE * 2);
    bf16_t* wg1b = (bf16_t*)take((size_t)E * D * 2);
    bf16_t* wg2b = (bf16_t*)take((size_t)D * E * 2);
    bf16_t* g1b  = (bf16_t*)take((size_t)M * D * 2);
    bf16_t* hb   = xb;      // x consumed before h written
    bf16_t* gtb  = qb;      // q consumed before gate written

    // 1) precision conversion
    tnl_cvt_bf16<<<(int)((MNE + 255) / 256), 256, 0, stream>>>(x, xb, (int)MNE);
    tnl_cvt_bf16<<<(int)((EE + 255) / 256), 256, 0, stream>>>(Wq, wqb, (int)EE);
    tnl_cvt_bf16<<<(int)((EE + 255) / 256), 256, 0, stream>>>(Wk, wkb, (int)EE);
    tnl_cvt_bf16<<<(int)((EE + 255) / 256), 256, 0, stream>>>(Wv, wvb, (int)EE);
    tnl_cvt_bf16<<<(int)((EE + 255) / 256), 256, 0, stream>>>(Wo, wob, (int)EE);
    tnl_cvt_bf16<<<(E * D + 255) / 256, 256, 0, stream>>>(Wg1, wg1b, E * D);
    tnl_cvt_bf16<<<(D * E + 255) / 256, 256, 0, stream>>>(Wg2, wg2b, D * E);

    // 2) q/k/v = silu(x@W)  and  g1 = x@Wg1
    dim3 gEE(E / 128, M / 128);
    tnl_gemm_bf16<128, 1, bf16_t><<<gEE, 256, 0, stream>>>(xb, wqb, qb, M, E, E);
    tnl_gemm_bf16<128, 1, bf16_t><<<gEE, 256, 0, stream>>>(xb, wkb, kb, M, E, E);
    tnl_gemm_bf16<128, 1, bf16_t><<<gEE, 256, 0, stream>>>(xb, wvb, vb, M, E, E);
    tnl_gemm_bf16<64, 0, bf16_t><<<dim3(1, M / 128), 256, 0, stream>>>(xb, wg1b, g1b, M, D, E);

    // 3) lightning attention (TDM-staged Q/K)
    tnl_attn<<<dim3(16, 2), 256, 0, stream>>>(qb, kb, vb, obuf);

    // 4) gate = sigmoid(g1 @ Wg2)   (reuses q buffer)
    tnl_gemm_bf16<128, 2, bf16_t><<<gEE, 256, 0, stream>>>(g1b, wg2b, gtb, M, E, D);

    // 5) h = (LN(o)*gamma+beta) * gate   (reuses x buffer)
    tnl_ln_gate<<<M, 256, 0, stream>>>(obuf, gamma, beta, gtb, hb);

    // 6) out = h @ Wo  (fp32)
    tnl_gemm_bf16<128, 0, float><<<gEE, 256, 0, stream>>>(hb, wob, out, M, E, E);
}
